// AllSetConv_47553877901911
// MI455X (gfx1250) — matrix operand order repaired and checked by reading, not compiled, not run on hardware
//
#include <hip/hip_runtime.h>
#include <hip/hip_bf16.h>

typedef __attribute__((ext_vector_type(16))) _Float16 v16h;
typedef __attribute__((ext_vector_type(8)))  float    v8f;

#define DIM 128
#define LN_EPS 1e-5f

// ---------- WMMA fragment helpers (CDNA5 16x16x32 f16, wave32) ----------

// A fragment from an LDS tile sA[16][DIM] (row-major f16).
// Lane L<16: row=L,   elems 0..15 = K k0+0..15
// Lane L>=16: row=L-16, elems 0..15 = K k0+16..31
__device__ __forceinline__ v16h load_a_frag(const _Float16* sA, int lane, int k0) {
  int m  = lane & 15;
  int kh = (lane >> 4) * 16;
  const _Float16* base = sA + m * DIM + k0 + kh;
  v16h a;
#pragma unroll
  for (int e = 0; e < 8; ++e) {
    a[e]     = base[e];
    a[e + 8] = base[8 + e];
  }
  return a;
}

// B fragment from a pre-transposed weight wt[N][K] (f16).
// Lane L<16: col N=n0+L, elems 0..15 = K k0+0..15 (contiguous 32B)
// Lane L>=16: col N=n0+L-16, elems 0..15 = K k0+16..31 (contiguous 32B)
__device__ __forceinline__ v16h load_b_frag(const _Float16* __restrict__ wt,
                                            int lane, int n0, int k0) {
  int n  = n0 + (lane & 15);
  int kk = k0 + (lane >> 4) * 16;
  return *(const v16h*)(wt + n * DIM + kk);
}

__device__ __forceinline__ v8f gemm_tile(const _Float16* sA,
                                         const _Float16* __restrict__ wt,
                                         int lane, int n0) {
  v8f c = {};
#pragma unroll
  for (int k0 = 0; k0 < DIM; k0 += 32) {
    v16h a = load_a_frag(sA, lane, k0);
    v16h b = load_b_frag(wt, lane, n0, k0);
    c = __builtin_amdgcn_wmma_f32_16x16x32_f16(false, a, false, b,
                                               (short)0, c, false, false);
  }
  return c;
}

// ---------- weight prep: f32 [K][N] -> f16 transposed [N][K] ----------
__global__ __launch_bounds__(DIM) void prep_wt(const float* __restrict__ w,
                                               _Float16* __restrict__ wt) {
  int k = blockIdx.x;   // 0..127
  int n = threadIdx.x;  // 0..127
  wt[n * DIM + k] = (_Float16)w[k * DIM + n];
}

__global__ void zero_f32(float* __restrict__ p, int n) {
  int i = blockIdx.x * blockDim.x + threadIdx.x;
  if (i < n) p[i] = 0.0f;
}

// ---------- fused encoder: hw = (relu(LN(relu(xW1+b1))W2+b2)) @ Wc ----------
__global__ __launch_bounds__(256) void enc_fused(
    const float* __restrict__ x,
    const _Float16* __restrict__ wt1, const float* __restrict__ b1,
    const float* __restrict__ g,  const float* __restrict__ beta,
    const _Float16* __restrict__ wt2, const float* __restrict__ b2,
    const _Float16* __restrict__ wtc,
    float* __restrict__ hw, int nrows) {
  __shared__ _Float16 sA[16 * DIM];     // 4 KB  (current GEMM A operand, f16)
  __shared__ float    sT[16 * DIM];     // 8 KB  (f32 staging for LayerNorm)
  __shared__ float    sPart[16][16], sPart2[16][16];
  __shared__ float    sMean[16], sRstd[16];

  const int tid   = threadIdx.x;
  const int lane  = tid & 31;
  const int wave  = tid >> 5;        // 0..7 -> N tile
  const int n0    = wave * 16;
  const int row0  = blockIdx.x * 16;
  const int khalf = lane >> 4;
  const int ncol  = n0 + (lane & 15);

  // load x tile -> f16 A
  for (int i = tid; i < 16 * DIM; i += 256) {
    int r = i >> 7, c = i & (DIM - 1);
    float v = (row0 + r < nrows) ? x[(size_t)(row0 + r) * DIM + c] : 0.0f;
    sA[r * DIM + c] = (_Float16)v;
  }
  __syncthreads();

  // GEMM1 + bias + relu -> sT
  v8f c1 = gemm_tile(sA, wt1, lane, n0);
  {
    float bias = b1[ncol];
#pragma unroll
    for (int r = 0; r < 8; ++r) {
      float v = c1[r] + bias;
      sT[(r + 8 * khalf) * DIM + ncol] = v > 0.0f ? v : 0.0f;
    }
  }
  __syncthreads();

  // LayerNorm over D=128 per row
  {
    int r = tid >> 4, c = tid & 15;
    float s = 0.0f, s2 = 0.0f;
#pragma unroll
    for (int j = 0; j < 8; ++j) {
      float v = sT[r * DIM + c + 16 * j];
      s += v; s2 += v * v;
    }
    sPart[r][c] = s; sPart2[r][c] = s2;
  }
  __syncthreads();
  if (tid < 16) {
    float s = 0.0f, s2 = 0.0f;
    for (int c = 0; c < 16; ++c) { s += sPart[tid][c]; s2 += sPart2[tid][c]; }
    float m   = s * (1.0f / DIM);
    float var = s2 * (1.0f / DIM) - m * m;
    sMean[tid] = m;
    sRstd[tid] = rsqrtf(var + LN_EPS);
  }
  __syncthreads();
  {
    int r = tid >> 4, c = tid & 15;
    float m = sMean[r], rs = sRstd[r];
#pragma unroll
    for (int j = 0; j < 8; ++j) {
      int col = c + 16 * j;
      float v = (sT[r * DIM + col] - m) * rs * g[col] + beta[col];
      sA[r * DIM + col] = (_Float16)v;
    }
  }
  __syncthreads();

  // GEMM2 + bias + relu -> h (back into sA as f16)
  v8f c2 = gemm_tile(sA, wt2, lane, n0);
  __syncthreads();   // all waves done reading sA before overwrite
  {
    float bias = b2[ncol];
#pragma unroll
    for (int r = 0; r < 8; ++r) {
      float v = c2[r] + bias;
      sA[(r + 8 * khalf) * DIM + ncol] = (_Float16)(v > 0.0f ? v : 0.0f);
    }
  }
  __syncthreads();

  // GEMM3 (conv weight, no bias) -> hw (f32, global)
  v8f c3 = gemm_tile(sA, wtc, lane, n0);
#pragma unroll
  for (int r = 0; r < 8; ++r) {
    int row = row0 + r + 8 * khalf;
    if (row < nrows) hw[(size_t)row * DIM + ncol] = c3[r];
  }
}

// ---------- scatter: agg[row] += val * hw[col], rowsum[row] += val ----------
// One wave per nonzero: 32 lanes x float4 = 512B coalesced gather per row.
__global__ __launch_bounds__(256) void scatter_msgs(
    const int* __restrict__ rows, const int* __restrict__ cols,
    const float* __restrict__ vals, const float* __restrict__ hw,
    float* __restrict__ agg, float* __restrict__ rowsum, int nnz) {
  int w    = (blockIdx.x << 3) + (threadIdx.x >> 5);  // 8 waves / block
  int lane = threadIdx.x & 31;
  if (w >= nnz) return;
  int   r = rows[w];
  int   c = cols[w];
  float v = vals[w];
  const float4 d = *((const float4*)(hw + (size_t)c * DIM) + lane);
  float* dst = agg + (size_t)r * DIM + lane * 4;
  atomicAdd(dst + 0, d.x * v);
  atomicAdd(dst + 1, d.y * v);
  atomicAdd(dst + 2, d.z * v);
  atomicAdd(dst + 3, d.w * v);
  if (lane == 0) atomicAdd(rowsum + r, v);
}

// ---------- fused decoder: out = relu(LN(relu((agg/rowsum)W1+b1))W2+b2) ----------
__global__ __launch_bounds__(256) void dec_fused(
    const float* __restrict__ agg, const float* __restrict__ rowsum,
    const _Float16* __restrict__ wt1, const float* __restrict__ b1,
    const float* __restrict__ g,  const float* __restrict__ beta,
    const _Float16* __restrict__ wt2, const float* __restrict__ b2,
    float* __restrict__ out, int nrows) {
  __shared__ _Float16 sA[16 * DIM];
  __shared__ float    sT[16 * DIM];
  __shared__ float    sPart[16][16], sPart2[16][16];
  __shared__ float    sMean[16], sRstd[16];
  __shared__ float    sInv[16];

  const int tid   = threadIdx.x;
  const int lane  = tid & 31;
  const int wave  = tid >> 5;
  const int n0    = wave * 16;
  const int row0  = blockIdx.x * 16;
  const int khalf = lane >> 4;
  const int ncol  = n0 + (lane & 15);

  if (tid < 16) {
    float rs = (row0 + tid < nrows) ? rowsum[row0 + tid] : 1.0f;
    sInv[tid] = 1.0f / rs;
  }
  __syncthreads();
  for (int i = tid; i < 16 * DIM; i += 256) {
    int r = i >> 7, c = i & (DIM - 1);
    float v = (row0 + r < nrows) ? agg[(size_t)(row0 + r) * DIM + c] * sInv[r] : 0.0f;
    sA[r * DIM + c] = (_Float16)v;
  }
  __syncthreads();

  // GEMM1 + bias + relu -> sT
  v8f c1 = gemm_tile(sA, wt1, lane, n0);
  {
    float bias = b1[ncol];
#pragma unroll
    for (int r = 0; r < 8; ++r) {
      float v = c1[r] + bias;
      sT[(r + 8 * khalf) * DIM + ncol] = v > 0.0f ? v : 0.0f;
    }
  }
  __syncthreads();
  {
    int r = tid >> 4, c = tid & 15;
    float s = 0.0f, s2 = 0.0f;
#pragma unroll
    for (int j = 0; j < 8; ++j) {
      float v = sT[r * DIM + c + 16 * j];
      s += v; s2 += v * v;
    }
    sPart[r][c] = s; sPart2[r][c] = s2;
  }
  __syncthreads();
  if (tid < 16) {
    float s = 0.0f, s2 = 0.0f;
    for (int c = 0; c < 16; ++c) { s += sPart[tid][c]; s2 += sPart2[tid][c]; }
    float m   = s * (1.0f / DIM);
    float var = s2 * (1.0f / DIM) - m * m;
    sMean[tid] = m;
    sRstd[tid] = rsqrtf(var + LN_EPS);
  }
  __syncthreads();
  {
    int r = tid >> 4, c = tid & 15;
    float m = sMean[r], rs = sRstd[r];
#pragma unroll
    for (int j = 0; j < 8; ++j) {
      int col = c + 16 * j;
      float v = (sT[r * DIM + col] - m) * rs * g[col] + beta[col];
      sA[r * DIM + col] = (_Float16)v;
    }
  }
  __syncthreads();

  // GEMM2 + bias + final relu -> out
  v8f c2 = gemm_tile(sA, wt2, lane, n0);
  {
    float bias = b2[ncol];
#pragma unroll
    for (int r = 0; r < 8; ++r) {
      int row = row0 + r + 8 * khalf;
      if (row < nrows) {
        float v = c2[r] + bias;
        out[(size_t)row * DIM + ncol] = v > 0.0f ? v : 0.0f;
      }
    }
  }
}

// ------------------------------- launch -------------------------------
extern "C" void kernel_launch(void* const* d_in, const int* in_sizes, int n_in,
                              void* d_out, int out_size, void* d_ws, size_t ws_size,
                              hipStream_t stream) {
  const float* x        = (const float*)d_in[0];
  const int*   inc_rows = (const int*)d_in[1];
  const int*   inc_cols = (const int*)d_in[2];
  const float* inc_vals = (const float*)d_in[3];
  const float* enc_w1 = (const float*)d_in[4];
  const float* enc_b1 = (const float*)d_in[5];
  const float* enc_g  = (const float*)d_in[6];
  const float* enc_be = (const float*)d_in[7];
  const float* enc_w2 = (const float*)d_in[8];
  const float* enc_b2 = (const float*)d_in[9];
  const float* conv_w = (const float*)d_in[10];
  const float* dec_w1 = (const float*)d_in[11];
  const float* dec_b1 = (const float*)d_in[12];
  const float* dec_g  = (const float*)d_in[13];
  const float* dec_be = (const float*)d_in[14];
  const float* dec_w2 = (const float*)d_in[15];
  const float* dec_b2 = (const float*)d_in[16];

  const int nsrc = in_sizes[0] / DIM;   // 100000
  const int nnz  = in_sizes[1];         // 1600000
  const int ntgt = out_size / DIM;      // 20000

  // workspace layout (256B aligned pieces)
  char* ws = (char*)d_ws;
  _Float16* wt_e1 = (_Float16*)ws;  ws += DIM * DIM * 2;
  _Float16* wt_e2 = (_Float16*)ws;  ws += DIM * DIM * 2;
  _Float16* wt_cv = (_Float16*)ws;  ws += DIM * DIM * 2;
  _Float16* wt_d1 = (_Float16*)ws;  ws += DIM * DIM * 2;
  _Float16* wt_d2 = (_Float16*)ws;  ws += DIM * DIM * 2;
  float* hw     = (float*)ws;       ws += (size_t)nsrc * DIM * 4;
  float* agg    = (float*)ws;       ws += (size_t)ntgt * DIM * 4;
  float* rowsum = (float*)ws;       // ntgt floats, contiguous after agg

  // 1) weight transpose + f16 convert
  prep_wt<<<DIM, DIM, 0, stream>>>(enc_w1, wt_e1);
  prep_wt<<<DIM, DIM, 0, stream>>>(enc_w2, wt_e2);
  prep_wt<<<DIM, DIM, 0, stream>>>(conv_w, wt_cv);
  prep_wt<<<DIM, DIM, 0, stream>>>(dec_w1, wt_d1);
  prep_wt<<<DIM, DIM, 0, stream>>>(dec_w2, wt_d2);

  // 2) zero accumulators (agg + rowsum are contiguous)
  {
    int nz = ntgt * (DIM + 1);
    zero_f32<<<(nz + 255) / 256, 256, 0, stream>>>(agg, nz);
  }

  // 3) fused encoder -> hw
  enc_fused<<<(nsrc + 15) / 16, 256, 0, stream>>>(
      x, wt_e1, enc_b1, enc_g, enc_be, wt_e2, enc_b2, wt_cv, hw, nsrc);

  // 4) scatter: one wave per nonzero
  scatter_msgs<<<(nnz + 7) / 8, 256, 0, stream>>>(
      inc_rows, inc_cols, inc_vals, hw, agg, rowsum, nnz);

  // 5) fused decoder -> d_out
  dec_fused<<<(ntgt + 15) / 16, 256, 0, stream>>>(
      agg, rowsum, wt_d1, dec_b1, dec_g, dec_be, wt_d2, dec_b2,
      (float*)d_out, ntgt);
}
